// HeterogeneousGCN_70480413327755
// MI455X (gfx1250) — matrix operand order repaired and checked by reading, not compiled, not run on hardware
//
#include <hip/hip_runtime.h>

// ---------------------------------------------------------------------------
// HeterogeneousGCN forward for MI455X (gfx1250, wave32, WMMA).
//
// All dense [M,K]@[K,64] products go through v_wmma_f32_16x16x32_bf16
// (bf16 A/B, fp32 accumulate). Weights are repacked fp32->bf16 into LDS in
// contiguous k order, k = 16*(lane/16) + e, matching A (one 32B ds_load_b128 pair per tile).
// A is fetched with b128 global loads matching the B packing's contiguous k order (k = 16*(lane/16) + e),
// optionally scaled per-row by 1/max(count,1) so the SAGE mean
// division fuses into the GEMM (removes a full read+write pass over every
// aggregation buffer). Scatter-mean aggregation uses global_atomic_add_f32
// at float4 granularity (everything lives in the 192MB L2).
// ---------------------------------------------------------------------------

typedef __attribute__((ext_vector_type(16))) __bf16 v16bf;
typedef __attribute__((ext_vector_type(8)))  float  v8f;
typedef __attribute__((ext_vector_type(4)))  float  v4f;

#define FLAG_ACC  1   // C += result (otherwise overwrite)
#define FLAG_BIAS 2   // add bias[col]
#define FLAG_RELU 4   // relu on final store

__device__ __forceinline__ __bf16 f2bf(float f) {
    unsigned u = __builtin_bit_cast(unsigned, f);
    unsigned r = u + 0x7FFFu + ((u >> 16) & 1u);   // round-to-nearest-even
    return __builtin_bit_cast(__bf16, (unsigned short)(r >> 16));
}

// C[M,64] (+)= bf16(scale_r * A[M,K]) @ bf16(W[K,64]) (+ bias) (relu?)
// scale_r = 1/max(rowCnt[r],1) when rowCnt != null (fused SAGE mean).
// K in {64,128}. 4 waves / block, 16 rows per wave, 64 rows per block.
__global__ void __launch_bounds__(128)
gemm64_wmma(const float* __restrict__ A, const float* __restrict__ W,
            const float* __restrict__ bias, const float* __restrict__ rowCnt,
            float* __restrict__ C, int M, int K, int flags)
{
    // B operand, packed: [ktile][ntile][lane][16 bf16], lane holds
    // N = ntile*16 + (lane&15), elements e=0..15 -> K = ktile*32 + (lane<16?0:16) + e
    __shared__ __bf16 ldsB[8192];   // up to KT=4 (K=128): 4*4*32*16
    const int tid  = threadIdx.x;
    const int lane = tid & 31;
    const int wave = tid >> 5;
    const int KT   = K >> 5;

    for (int i = tid; i < (KT << 11); i += 128) {
        int e  = i & 15;
        int ln = (i >> 4) & 31;
        int nt = (i >> 9) & 3;
        int kt = i >> 11;
        int k  = (kt << 5) + ((ln >> 4) << 4) + e;
        int n  = (nt << 4) + (ln & 15);
        ldsB[i] = f2bf(W[k * 64 + n]);
    }
    __syncthreads();

    const int rowTile = blockIdx.x * 64 + wave * 16;
    int arow = rowTile + (lane & 15);
    if (arow >= M) arow = M - 1;          // tail rows: garbage fed, stores masked
    const int kgrp = (lane >> 4) << 4;     // A layout K-half select: 0 or 16
    const float rs = rowCnt ? (1.0f / fmaxf(rowCnt[arow], 1.0f)) : 1.0f;

    v8f acc0 = {0.f,0.f,0.f,0.f,0.f,0.f,0.f,0.f};
    v8f acc1 = acc0, acc2 = acc0, acc3 = acc0;

    for (int kt = 0; kt < KT; ++kt) {
        const float* ap = A + (size_t)arow * K + (kt << 5) + kgrp;
        v4f a0 = *(const v4f*)(ap) * rs;        // K_local kgrp+0..3
        v4f a1 = *(const v4f*)(ap + 4) * rs;    // kgrp+4..7
        v4f a2 = *(const v4f*)(ap + 8) * rs;    // kgrp+8..11
        v4f a3 = *(const v4f*)(ap + 12) * rs;   // kgrp+12..15
        v16bf a;
        #pragma unroll
        for (int j = 0; j < 4; ++j) {
            a[j]      = f2bf(a0[j]);
            a[j + 4]  = f2bf(a1[j]);
            a[j + 8]  = f2bf(a2[j]);
            a[j + 12] = f2bf(a3[j]);
        }
        const int lbase = ((kt << 2) * 32 + lane) << 4;
        v16bf b0 = *(const v16bf*)&ldsB[lbase];
        v16bf b1 = *(const v16bf*)&ldsB[lbase + 512];
        v16bf b2 = *(const v16bf*)&ldsB[lbase + 1024];
        v16bf b3 = *(const v16bf*)&ldsB[lbase + 1536];
        acc0 = __builtin_amdgcn_wmma_f32_16x16x32_bf16(false, a, false, b0, (short)0, acc0, false, false);
        acc1 = __builtin_amdgcn_wmma_f32_16x16x32_bf16(false, a, false, b1, (short)0, acc1, false, false);
        acc2 = __builtin_amdgcn_wmma_f32_16x16x32_bf16(false, a, false, b2, (short)0, acc2, false, false);
        acc3 = __builtin_amdgcn_wmma_f32_16x16x32_bf16(false, a, false, b3, (short)0, acc3, false, false);
    }

    // D layout: lane holds col (lane&15); VGPR g holds row g + (lane<16?0:8)
    const int col  = lane & 15;
    const int rofs = (lane >> 4) << 3;
    #pragma unroll
    for (int nt = 0; nt < 4; ++nt) {
        const v8f av = (nt == 0) ? acc0 : (nt == 1) ? acc1 : (nt == 2) ? acc2 : acc3;
        const int c  = (nt << 4) + col;
        const float bv = (flags & FLAG_BIAS) ? bias[c] : 0.0f;
        #pragma unroll
        for (int g = 0; g < 8; ++g) {
            int r = rowTile + rofs + g;
            if (r < M) {
                float v = av[g] + bv;
                float* p = C + (size_t)r * 64 + c;
                if (flags & FLAG_ACC)  v += *p;
                if (flags & FLAG_RELU) v = fmaxf(v, 0.0f);
                *p = v;
            }
        }
    }
}

__global__ void fill0_kernel(float* __restrict__ p, int n) {
    int i = blockIdx.x * blockDim.x + threadIdx.x;
    if (i < n) p[i] = 0.0f;
}

// one thread per (edge, 4-feature chunk): tot[dst[e]] += xsrc[src[e]]
__global__ void scatter_feat_kernel(const float* __restrict__ xsrc,
                                    const int* __restrict__ src,
                                    const int* __restrict__ dst,
                                    float* __restrict__ tot, int nE) {
    int gid = blockIdx.x * blockDim.x + threadIdx.x;
    int e = gid >> 4;
    if (e >= nE) return;
    int f = (gid & 15) << 2;
    const float* sp = xsrc + (size_t)src[e] * 64 + f;
    float*       tp = tot  + (size_t)dst[e] * 64 + f;
    v4f v = *(const v4f*)sp;                 // b128 gather
    atomicAdd(tp + 0, v.x);
    atomicAdd(tp + 1, v.y);
    atomicAdd(tp + 2, v.z);
    atomicAdd(tp + 3, v.w);
}

__global__ void count_kernel(const int* __restrict__ dst, float* __restrict__ cnt, int nE) {
    int e = blockIdx.x * blockDim.x + threadIdx.x;
    if (e < nE) atomicAdd(&cnt[dst[e]], 1.0f);
}

// out[i] = concat(xu[user_idx[i]], xj[job_idx[i]]) . Wp + bp
__global__ void pred_kernel(const float* __restrict__ xu, const float* __restrict__ xj,
                            const int* __restrict__ uidx, const int* __restrict__ jidx,
                            const float* __restrict__ Wp, const float* __restrict__ bp,
                            float* __restrict__ out, int B) {
    int i = blockIdx.x * blockDim.x + threadIdx.x;
    if (i >= B) return;
    const float* u = xu + (size_t)uidx[i] * 64;
    const float* j = xj + (size_t)jidx[i] * 64;
    float s = bp[0];
    #pragma unroll
    for (int k = 0; k < 64; k += 4) {
        v4f uv = *(const v4f*)(u + k);
        v4f wv = *(const v4f*)(Wp + k);
        s += uv.x * wv.x + uv.y * wv.y + uv.z * wv.z + uv.w * wv.w;
    }
    #pragma unroll
    for (int k = 0; k < 64; k += 4) {
        v4f jv = *(const v4f*)(j + k);
        v4f wv = *(const v4f*)(Wp + 64 + k);
        s += jv.x * wv.x + jv.y * wv.y + jv.z * wv.z + jv.w * wv.w;
    }
    out[i] = s;
}

extern "C" void kernel_launch(void* const* d_in, const int* in_sizes, int n_in,
                              void* d_out, int out_size, void* d_ws, size_t ws_size,
                              hipStream_t stream)
{
    (void)in_sizes; (void)n_in; (void)out_size; (void)ws_size;

    const float* x_user  = (const float*)d_in[0];
    const float* x_job   = (const float*)d_in[1];
    const float* x_skill = (const float*)d_in[2];
    const int* uj_src = (const int*)d_in[3];
    const int* uj_dst = (const int*)d_in[4];
    const int* ju_src = (const int*)d_in[5];
    const int* ju_dst = (const int*)d_in[6];
    const int* js_src = (const int*)d_in[7];
    const int* js_dst = (const int*)d_in[8];
    const int* sj_src = (const int*)d_in[9];
    const int* sj_dst = (const int*)d_in[10];
    const int* user_idx = (const int*)d_in[11];
    const int* job_idx  = (const int*)d_in[12];

    // params as jax pytree leaves (dicts flattened in sorted-key order):
    // emb: job(W,b), skill(W,b), user(W,b); layers[0..1]: js,ju,sj,uj each (Wl,bl,Wr); pred(W,b)
    const float* embJ_W = (const float*)d_in[13];
    const float* embJ_b = (const float*)d_in[14];
    const float* embS_W = (const float*)d_in[15];
    const float* embS_b = (const float*)d_in[16];
    const float* embU_W = (const float*)d_in[17];
    const float* embU_b = (const float*)d_in[18];
    struct Rel { const float *Wl, *bl, *Wr; };
    Rel L[2][4];  // [layer][js, ju, sj, uj]
    for (int l = 0; l < 2; ++l)
        for (int r = 0; r < 4; ++r) {
            int b = 19 + l * 12 + r * 3;
            L[l][r].Wl = (const float*)d_in[b];
            L[l][r].bl = (const float*)d_in[b + 1];
            L[l][r].Wr = (const float*)d_in[b + 2];
        }
    const float* predW = (const float*)d_in[43];
    const float* predb = (const float*)d_in[44];

    const int NU = 100000, NJ = 50000, NS = 5000;
    const int E_UJ = 250000, E_JS = 150000;

    float* ws = (float*)d_ws;
    size_t o = 0;
    auto take = [&](size_t n) { float* p = ws + o; o += n; return p; };
    float* U[2] = { take((size_t)NU * 64), take((size_t)NU * 64) };
    float* J[2] = { take((size_t)NJ * 64), take((size_t)NJ * 64) };
    float* S[2] = { take((size_t)NS * 64), take((size_t)NS * 64) };
    float* aggU = take((size_t)NU * 64);
    float* aggJ = take((size_t)NJ * 64);
    float* aggS = take((size_t)NS * 64);
    float* cntU = take(NU);
    float* cntJ = take(NJ);
    float* cntS = take(NS);

    auto gemm = [&](const float* A, const float* W, const float* b, const float* rc,
                    float* Cm, int M, int K, int flags) {
        gemm64_wmma<<<(M + 63) / 64, 128, 0, stream>>>(A, W, b, rc, Cm, M, K, flags);
    };
    auto fill = [&](float* p, int n) {
        fill0_kernel<<<(n + 255) / 256, 256, 0, stream>>>(p, n);
    };
    // sum + degree count; mean division is fused into the consuming GEMM
    auto aggregate = [&](const float* xsrc, const int* src, const int* dst,
                         float* tot, float* cnt, int nE, int nDst) {
        fill(tot, nDst * 64);
        fill(cnt, nDst);
        scatter_feat_kernel<<<(nE * 16 + 255) / 256, 256, 0, stream>>>(xsrc, src, dst, tot, nE);
        count_kernel<<<(nE + 255) / 256, 256, 0, stream>>>(dst, cnt, nE);
    };

    // ---- embedding linears -------------------------------------------------
    gemm(x_user,  embU_W, embU_b, nullptr, U[0], NU, 128, FLAG_BIAS);
    gemm(x_job,   embJ_W, embJ_b, nullptr, J[0], NJ, 128, FLAG_BIAS);
    gemm(x_skill, embS_W, embS_b, nullptr, S[0], NS, 64,  FLAG_BIAS);

    // ---- 2 hetero SAGE layers (ping-pong U/J/S buffers) --------------------
    for (int l = 0; l < 2; ++l) {
        const int in = l, out = 1 - l;           // l=0: 0->1, l=1: 1->0
        const int relu = (l == 0) ? FLAG_RELU : 0;
        const Rel& js = L[l][0];
        const Rel& ju = L[l][1];
        const Rel& sj = L[l][2];
        const Rel& uj = L[l][3];

        // dst = job: relations uj then sj (HeteroConv sum), relu on last store
        aggregate(U[in], uj_src, uj_dst, aggJ, cntJ, E_UJ, NJ);
        gemm(aggJ,  uj.Wl, uj.bl,  cntJ,    J[out], NJ, 64, FLAG_BIAS);
        gemm(J[in], uj.Wr, nullptr, nullptr, J[out], NJ, 64, FLAG_ACC);
        aggregate(S[in], sj_src, sj_dst, aggJ, cntJ, E_JS, NJ);
        gemm(aggJ,  sj.Wl, sj.bl,  cntJ,    J[out], NJ, 64, FLAG_ACC | FLAG_BIAS);
        gemm(J[in], sj.Wr, nullptr, nullptr, J[out], NJ, 64, FLAG_ACC | relu);

        // dst = user: relation ju
        aggregate(J[in], ju_src, ju_dst, aggU, cntU, E_UJ, NU);
        gemm(aggU,  ju.Wl, ju.bl,  cntU,    U[out], NU, 64, FLAG_BIAS);
        gemm(U[in], ju.Wr, nullptr, nullptr, U[out], NU, 64, FLAG_ACC | relu);

        // dst = skill: relation js
        aggregate(J[in], js_src, js_dst, aggS, cntS, E_JS, NS);
        gemm(aggS,  js.Wl, js.bl,  cntS,    S[out], NS, 64, FLAG_BIAS);
        gemm(J[in], js.Wr, nullptr, nullptr, S[out], NS, 64, FLAG_ACC | relu);
    }

    // final features ended in buffer index 0
    pred_kernel<<<(8192 + 255) / 256, 256, 0, stream>>>(
        U[0], J[0], user_idx, job_idx, predW, predb, (float*)d_out, 8192);
}
